// RGATLayer_12962211299981
// MI455X (gfx1250) — matrix operand order repaired and checked by reading, not compiled, not run on hardware
//
#include <hip/hip_runtime.h>

// Problem constants (fixed by the reference)
#define N_NODES 100000
#define R_REL   4
#define E_EDGES 800000
#define IN_F    128
#define H_HEADS 4
#define DHEAD   32
#define OUT_F   128          // H_HEADS * DHEAD
#define NEG_SLOPE 0.2f

typedef __attribute__((ext_vector_type(16))) __bf16 v16bf;
typedef __attribute__((ext_vector_type(8)))  float  v8f;

// ---------------- helpers ----------------
__device__ __forceinline__ unsigned short f32_to_bf16_rne(float f) {
    unsigned u = __float_as_uint(f);
    unsigned r = u + 0x7fffu + ((u >> 16) & 1u);   // round-to-nearest-even
    return (unsigned short)(r >> 16);
}
__device__ __forceinline__ __bf16 bits_to_bf16(unsigned short u) {
    union { unsigned short s; __bf16 b; } x; x.s = u; return x.b;
}
// monotone float<->uint encoding for atomicMax-based segment_max
__device__ __forceinline__ unsigned enc_f32(float f) {
    unsigned b = __float_as_uint(f);
    return (b & 0x80000000u) ? ~b : (b | 0x80000000u);
}
__device__ __forceinline__ float dec_f32(unsigned u) {
    unsigned b = (u & 0x80000000u) ? (u & 0x7fffffffu) : ~u;
    return __uint_as_float(b);
}
#define ENC_NEG_INF 0x007FFFFFu   // enc(-inf)

__device__ __forceinline__ float lrelu(float z) {
    return z > 0.f ? z : NEG_SLOPE * z;
}

// ---------------- kernels ----------------
// split f32 -> bf16 hi/lo pair (for x)
__global__ void cvt_split_bf16_kernel(const float* __restrict__ in,
                                      unsigned short* __restrict__ hi,
                                      unsigned short* __restrict__ lo, int n) {
    int i = blockIdx.x * blockDim.x + threadIdx.x;
    if (i >= n) return;
    float f = in[i];
    unsigned short h = f32_to_bf16_rne(f);
    float hf = __uint_as_float(((unsigned)h) << 16);
    unsigned short l = f32_to_bf16_rne(f - hf);
    hi[i] = h; lo[i] = l;
}

// split + pack W into WMMA B-fragment order:
//   packed index i = ((((r*4 + t)*8 + w)*32 + l)*16 + e)
//   source: k = t*32 + (l>=16?16:0) + e ; col = w*16 + (l&15)
// so each lane's v16bf fragment is 32 contiguous bytes, coalesced across the wave.
__global__ void pack_w_kernel(const float* __restrict__ w_in,          // [R,128,128]
                              unsigned short* __restrict__ wh_pk,
                              unsigned short* __restrict__ wl_pk) {
    int i = blockIdx.x * blockDim.x + threadIdx.x;
    if (i >= R_REL * 128 * 128) return;
    int e = i & 15;
    int l = (i >> 4) & 31;
    int w = (i >> 9) & 7;
    int t = (i >> 12) & 3;
    int r = i >> 14;
    int k   = t * 32 + ((l >> 4) << 4) + e;
    int col = w * 16 + (l & 15);
    float f = w_in[(r * 128 + k) * 128 + col];
    unsigned short h = f32_to_bf16_rne(f);
    float hf = __uint_as_float(((unsigned)h) << 16);
    wh_pk[i] = h;
    wl_pk[i] = f32_to_bf16_rne(f - hf);
}

__global__ void zero_out_kernel(float* __restrict__ out) {
    int i = blockIdx.x * blockDim.x + threadIdx.x;
    if (i < N_NODES * OUT_F) out[i] = 0.f;
}

__global__ void init_rel_kernel(unsigned* __restrict__ emax_enc,
                                float* __restrict__ denom,
                                float* __restrict__ agg) {
    int i = blockIdx.x * blockDim.x + threadIdx.x;
    if (i < N_NODES * H_HEADS) { emax_enc[i] = ENC_NEG_INF; denom[i] = 0.f; }
    if (i < N_NODES * OUT_F)   agg[i] = 0.f;
}

// h = x @ W_r via split-bf16 WMMA. One block = 16 node rows; 8 waves = 8 col tiles.
#define LDS_STRIDE 132
__global__ void __launch_bounds__(256) gemm_wmma_kernel(
        const unsigned short* __restrict__ xh, const unsigned short* __restrict__ xl,
        const unsigned short* __restrict__ wh_pk, const unsigned short* __restrict__ wl_pk,
        float* __restrict__ hout) {
    __shared__ unsigned short sxh[16 * LDS_STRIDE];
    __shared__ unsigned short sxl[16 * LDS_STRIDE];
    const int row0 = blockIdx.x * 16;
    const int t = threadIdx.x;
    for (int idx = t; idx < 16 * 128; idx += 256) {
        int r = idx >> 7, c = idx & 127;
        sxh[r * LDS_STRIDE + c] = xh[(row0 + r) * 128 + c];
        sxl[r * LDS_STRIDE + c] = xl[(row0 + r) * 128 + c];
    }
    __syncthreads();

    const int wave = t >> 5;           // 0..7 -> 16-column tile
    const int lane = t & 31;
    const int col0 = wave * 16;
    const int m    = lane & 15;        // A row within tile
    const int half = lane >> 4;        // lane half
    const int ncol = lane & 15;        // B/D column within tile

    v8f acc = {};
    #pragma unroll
    for (int kt = 0; kt < 4; ++kt) {   // k0 = kt*32
        v16bf ah, alo;
        #pragma unroll
        for (int e = 0; e < 16; ++e) {
            // A layout: lane row m, k = half*16 + e
            int ka = kt * 32 + half * 16 + e;
            ah[e]  = bits_to_bf16(sxh[m * LDS_STRIDE + ka]);
            alo[e] = bits_to_bf16(sxl[m * LDS_STRIDE + ka]);
        }
        // B fragments: pre-packed, 32 contiguous bytes per lane, coalesced
        size_t fo = ((size_t)(kt * 8 + wave) * 32 + lane) * 16;
        v16bf bh  = *(const v16bf*)(wh_pk + fo);
        v16bf blo = *(const v16bf*)(wl_pk + fo);
        acc = __builtin_amdgcn_wmma_f32_16x16x32_bf16(false, ah,  false, bh,  (short)0, acc, false, false);
        acc = __builtin_amdgcn_wmma_f32_16x16x32_bf16(false, ah,  false, blo, (short)0, acc, false, false);
        acc = __builtin_amdgcn_wmma_f32_16x16x32_bf16(false, alo, false, bh,  (short)0, acc, false, false);
    }
    // D layout: lane ncol = column, VGPR v -> row v + 8*half
    #pragma unroll
    for (int v = 0; v < 8; ++v) {
        int row = row0 + v + half * 8;
        hout[row * 128 + col0 + ncol] = acc[v];
    }
}

// el[n,h] = h[n,h,:].al[h,:]  ;  er likewise (float4 vectorized)
__global__ void elr_kernel(const float* __restrict__ h,
                           const float* __restrict__ al, const float* __restrict__ ar,
                           float* __restrict__ el, float* __restrict__ er) {
    int i = blockIdx.x * blockDim.x + threadIdx.x;   // n*H + head
    if (i >= N_NODES * H_HEADS) return;
    int n = i >> 2, head = i & 3;
    const float4* hp  = (const float4*)(h  + (size_t)n * 128 + head * DHEAD);
    const float4* alp = (const float4*)(al + head * DHEAD);
    const float4* arp = (const float4*)(ar + head * DHEAD);
    float sl = 0.f, sr = 0.f;
    #pragma unroll
    for (int d = 0; d < DHEAD / 4; ++d) {
        float4 v = hp[d], a = alp[d], b = arp[d];
        sl += v.x * a.x + v.y * a.y + v.z * a.z + v.w * a.w;
        sr += v.x * b.x + v.y * b.y + v.z * b.z + v.w * b.w;
    }
    el[i] = sl; er[i] = sr;
}

// pass 1: segment_max over dst via encoded atomicMax
__global__ void edge_max_kernel(const int* __restrict__ src, const int* __restrict__ dst,
                                const float* __restrict__ el, const float* __restrict__ er,
                                unsigned* __restrict__ emax_enc) {
    int i = blockIdx.x * blockDim.x + threadIdx.x;
    if (i >= E_EDGES * H_HEADS) return;
    int e = i >> 2, head = i & 3;
    int s = src[e], d = dst[e];
    float z = lrelu(el[s * 4 + head] + er[d * 4 + head]);
    atomicMax(&emax_enc[d * 4 + head], enc_f32(z));
}

// pass 2: ee = exp(z - emax[dst]) stored per edge-head; denom = segment_sum(ee)
__global__ void edge_sum_kernel(const int* __restrict__ src, const int* __restrict__ dst,
                                const float* __restrict__ el, const float* __restrict__ er,
                                const unsigned* __restrict__ emax_enc,
                                float* __restrict__ ee_out,
                                float* __restrict__ denom) {
    int i = blockIdx.x * blockDim.x + threadIdx.x;
    if (i >= E_EDGES * H_HEADS) return;
    int e = i >> 2, head = i & 3;
    int s = src[e], d = dst[e];
    float z  = lrelu(el[s * 4 + head] + er[d * 4 + head]);
    float ee = __expf(z - dec_f32(emax_enc[d * 4 + head]));
    ee_out[i] = ee;
    atomicAdd(&denom[d * 4 + head], ee);
}

// pass 3: agg[dst] += (ee/denom[dst]) * h[src]   (one thread per edge*feature)
__global__ void edge_agg_kernel(const int* __restrict__ src, const int* __restrict__ dst,
                                const float* __restrict__ ee,
                                const float* __restrict__ denom,
                                const float* __restrict__ h,
                                float* __restrict__ agg) {
    int i = blockIdx.x * blockDim.x + threadIdx.x;   // E*128 = 102.4M < 2^31
    if (i >= E_EDGES * OUT_F) return;
    int e = i >> 7, c = i & 127, head = c >> 5;
    int s = src[e], d = dst[e];
    float alpha = ee[e * 4 + head] / denom[d * 4 + head];
    atomicAdd(&agg[d * 128 + c], alpha * h[s * 128 + c]);
}

// out += relu(agg_r)   (sum of relus is >= 0, so the outer relu is identity)
__global__ void accum_kernel(const float* __restrict__ agg, float* __restrict__ out) {
    int i = blockIdx.x * blockDim.x + threadIdx.x;
    if (i < N_NODES * OUT_F) out[i] += fmaxf(agg[i], 0.f);
}

// ---------------- launch ----------------
extern "C" void kernel_launch(void* const* d_in, const int* in_sizes, int n_in,
                              void* d_out, int out_size, void* d_ws, size_t ws_size,
                              hipStream_t stream) {
    const float* x      = (const float*)d_in[0];
    const int*   src    = (const int*)d_in[1];
    const int*   dst    = (const int*)d_in[2];
    const float* fc_w   = (const float*)d_in[3];
    const float* attn_l = (const float*)d_in[4];
    const float* attn_r = (const float*)d_in[5];
    float* out = (float*)d_out;

    char* ws = (char*)d_ws;
    size_t off = 0;
    auto alloc = [&](size_t bytes) -> char* {
        char* p = ws + off;
        off = (off + bytes + 255) & ~(size_t)255;
        return p;
    };
    float*          hbuf  = (float*)alloc((size_t)N_NODES * 128 * 4);
    float*          agg   = (float*)alloc((size_t)N_NODES * 128 * 4);
    float*          el    = (float*)alloc((size_t)N_NODES * H_HEADS * 4);
    float*          er    = (float*)alloc((size_t)N_NODES * H_HEADS * 4);
    unsigned*       emax  = (unsigned*)alloc((size_t)N_NODES * H_HEADS * 4);
    float*          denom = (float*)alloc((size_t)N_NODES * H_HEADS * 4);
    float*          eebuf = (float*)alloc((size_t)E_EDGES * H_HEADS * 4);
    unsigned short* xh    = (unsigned short*)alloc((size_t)N_NODES * 128 * 2);
    unsigned short* xl    = (unsigned short*)alloc((size_t)N_NODES * 128 * 2);
    unsigned short* whpk  = (unsigned short*)alloc((size_t)R_REL * 128 * 128 * 2);
    unsigned short* wlpk  = (unsigned short*)alloc((size_t)R_REL * 128 * 128 * 2);

    const int B = 256;
    int n_x = N_NODES * 128;
    int n_w = R_REL * 128 * 128;
    cvt_split_bf16_kernel<<<(n_x + B - 1) / B, B, 0, stream>>>(x, xh, xl, n_x);
    pack_w_kernel<<<(n_w + B - 1) / B, B, 0, stream>>>(fc_w, whpk, wlpk);
    zero_out_kernel<<<(N_NODES * OUT_F + B - 1) / B, B, 0, stream>>>(out);

    const int n_nf = N_NODES * OUT_F;                // 12.8M
    const int n_eh = E_EDGES * H_HEADS;              // 3.2M
    const int n_ef = E_EDGES * OUT_F;                // 102.4M

    for (int r = 0; r < R_REL; ++r) {
        init_rel_kernel<<<(n_nf + B - 1) / B, B, 0, stream>>>(emax, denom, agg);
        gemm_wmma_kernel<<<N_NODES / 16, 256, 0, stream>>>(
            xh, xl, whpk + (size_t)r * 128 * 128, wlpk + (size_t)r * 128 * 128, hbuf);
        elr_kernel<<<(N_NODES * H_HEADS + B - 1) / B, B, 0, stream>>>(
            hbuf, attn_l + (size_t)r * OUT_F, attn_r + (size_t)r * OUT_F, el, er);
        edge_max_kernel<<<(n_eh + B - 1) / B, B, 0, stream>>>(
            src + (size_t)r * E_EDGES, dst + (size_t)r * E_EDGES, el, er, emax);
        edge_sum_kernel<<<(n_eh + B - 1) / B, B, 0, stream>>>(
            src + (size_t)r * E_EDGES, dst + (size_t)r * E_EDGES, el, er, emax, eebuf, denom);
        edge_agg_kernel<<<(n_ef + B - 1) / B, B, 0, stream>>>(
            src + (size_t)r * E_EDGES, dst + (size_t)r * E_EDGES, eebuf, denom, hbuf, agg);
        accum_kernel<<<(n_nf + B - 1) / B, B, 0, stream>>>(agg, out);
    }
}
